// BeamGenerator_28492813041966
// MI455X (gfx1250) — compile-verified, matrix operand order checked by reading, not run on hardware
//
#include <hip/hip_runtime.h>

#define VOCABN  50257
#define VPAD    50432   // padded to multiple of 256 for 256-wide N blocks
#define DD      512
#define BSZ     32
#define SRCLEN  128
#define NROWS   128     // BSZ * BEAM
#define MAXTGT  48
#define TOKW    50      // MAXTGT + 2
#define SCW     49      // MAXTGT + 1
#define PAD_T   1
#define EOS_T   2
#define NEGINF  (-1e30f)

typedef __attribute__((ext_vector_type(16))) __bf16 v16bf;
typedef __attribute__((ext_vector_type(8)))  float  v8f;

union BfFrag { v16bf v; uint4 q[2]; };

__device__ __forceinline__ __bf16 f2bf(float f) {
  union { unsigned u; float f; } in; in.f = f;
  unsigned r = in.u + 0x7fffu + ((in.u >> 16) & 1u);   // round-to-nearest-even
  union { unsigned short s; __bf16 b; } out; out.s = (unsigned short)(r >> 16);
  return out.b;
}

// ---------------- prep: fp32 -> bf16 conversions ----------------
__global__ __launch_bounds__(256) void conv_embed(const float* __restrict__ e,
                                                  __bf16* __restrict__ o) {
  size_t i = (size_t)blockIdx.x * 256 + threadIdx.x;
  if (i >= (size_t)VPAD * DD) return;
  size_t row = i >> 9;
  float v = (row < VOCABN) ? e[i] : 0.f;
  o[i] = f2bf(v);
}

__global__ __launch_bounds__(256) void conv_transpose(const float* __restrict__ W,
                                                      __bf16* __restrict__ o) {
  int n = blockIdx.x;
  int k = blockIdx.y * 256 + threadIdx.x;
  o[(size_t)n * DD + k] = f2bf(W[(size_t)k * DD + n]);   // o[n][k] = W[k][n]
}

// ---------------- unified WMMA GEMM with tanh epilogue ----------------
// A rows gathered from ebf by token id (or fixedTok), staged via LDS.
// B stored [N,K] bf16. out = tanh(A @ B^T(+bias)(+ctx)); fp32 and/or bf16 out.
__global__ __launch_bounds__(128) void emb_gemm_tanh(
    const int* __restrict__ tok, int tokStride, int tokOff, int fixedTok,
    const __bf16* __restrict__ ebf,
    const __bf16* __restrict__ Bt,
    const float* __restrict__ bias,
    const float* __restrict__ ctxAdd, int ctxShift,
    float* __restrict__ outF, __bf16* __restrict__ outBf, int Nout)
{
  __shared__ __bf16 lA[16 * DD];           // 16 KB staged A tile
  __shared__ int    stok[16];

  const int lane  = threadIdx.x & 31;
  const int wave  = threadIdx.x >> 5;
  const int mbase = blockIdx.y * 16;
  const int nbase = blockIdx.x * 64 + wave * 16;
  const int r     = lane & 15;
  const int hi    = (lane >> 4) ? 8 : 0;

  if (threadIdx.x < 16)
    stok[threadIdx.x] = tok ? tok[(mbase + threadIdx.x) * tokStride + tokOff] : fixedTok;
  __syncthreads();
  {
    uint4* ldst = (uint4*)lA;
    for (int i = threadIdx.x; i < (16 * DD) / 8; i += 128) {
      int rr = i >> 6, seg = i & 63;
      ldst[i] = ((const uint4*)(ebf + (size_t)stok[rr] * DD))[seg];
    }
  }
  __syncthreads();

  const __bf16* aL   = lA + r * DD + hi;
  const __bf16* bptr = Bt + (size_t)(nbase + r) * DD + hi;

  v8f acc = {};
  #pragma unroll 4
  for (int k = 0; k < DD; k += 32) {
    BfFrag fa, fb;
    fa.q[0] = *reinterpret_cast<const uint4*>(aL + k);
    fa.q[1] = *reinterpret_cast<const uint4*>(aL + k + 16);
    fb.q[0] = *reinterpret_cast<const uint4*>(bptr + k);
    fb.q[1] = *reinterpret_cast<const uint4*>(bptr + k + 16);
    acc = __builtin_amdgcn_wmma_f32_16x16x32_bf16(false, fa.v, false, fb.v,
                                                  (short)0, acc, false, false);
  }

  const int col  = nbase + (lane & 15);
  const int rowb = mbase + ((lane >> 4) ? 8 : 0);
  if (col < Nout) {
    const float bv = bias ? bias[col] : 0.f;
    #pragma unroll
    for (int j = 0; j < 8; ++j) {
      int row = rowb + j;
      float v = acc[j] + bv;
      if (ctxAdd) v += ctxAdd[(size_t)(row >> ctxShift) * DD + col];
      v = tanhf(v);
      if (outF)  outF [(size_t)row * Nout + col] = v;
      if (outBf) outBf[(size_t)row * Nout + col] = f2bf(v);
    }
  }
}

// ---------------- logits: h(M x 512) @ embed^T -> [M, VOCAB] fp32 --------------
// Block tile: 16 M x 256 N. A tile staged in LDS (shared by 4 waves);
// each wave computes 4 N-tiles, reusing the A fragment 4x per k-step.
__global__ __launch_bounds__(128) void logits_gemm(
    const __bf16* __restrict__ hbf, const __bf16* __restrict__ ebf,
    float* __restrict__ out, int NV)
{
  __shared__ __bf16 lA[16 * DD];           // 16 KB staged A tile

  const int lane  = threadIdx.x & 31;
  const int wave  = threadIdx.x >> 5;
  const int mbase = blockIdx.y * 16;
  const int nbase = blockIdx.x * 256 + wave * 64;
  const int r     = lane & 15;
  const int hi    = (lane >> 4) ? 8 : 0;

  {
    const uint4* gsrc = (const uint4*)(hbf + (size_t)mbase * DD);
    uint4* ldst = (uint4*)lA;
    for (int i = threadIdx.x; i < (16 * DD) / 8; i += 128) ldst[i] = gsrc[i];
  }
  __syncthreads();

  const __bf16* aL = lA + r * DD + hi;
  const __bf16* bp = ebf + (size_t)(nbase + r) * DD + hi;   // tile t at +t*16*DD

  v8f acc[4] = {{}, {}, {}, {}};
  #pragma unroll 2
  for (int k = 0; k < DD; k += 32) {
    __builtin_prefetch((const void*)(bp + k + 256), 0, 1);
    BfFrag fa;
    fa.q[0] = *reinterpret_cast<const uint4*>(aL + k);
    fa.q[1] = *reinterpret_cast<const uint4*>(aL + k + 16);
    #pragma unroll
    for (int t = 0; t < 4; ++t) {
      const __bf16* bt = bp + (size_t)t * 16 * DD;
      BfFrag fb;
      fb.q[0] = *reinterpret_cast<const uint4*>(bt + k);
      fb.q[1] = *reinterpret_cast<const uint4*>(bt + k + 16);
      acc[t] = __builtin_amdgcn_wmma_f32_16x16x32_bf16(false, fa.v, false, fb.v,
                                                       (short)0, acc[t], false, false);
    }
  }

  const int rowb = mbase + ((lane >> 4) ? 8 : 0);
  #pragma unroll
  for (int t = 0; t < 4; ++t) {
    const int col = nbase + t * 16 + (lane & 15);
    if (col < NV) {
      #pragma unroll
      for (int j = 0; j < 8; ++j)
        out[(size_t)(rowb + j) * NV + col] = acc[t][j];
    }
  }
}

// ---------------- ctx = mean over src positions ----------------
__global__ __launch_bounds__(256) void ctx_mean(const float* __restrict__ enc,
                                                float* __restrict__ ctx) {
  int b = blockIdx.x;
  int d = blockIdx.y * 256 + threadIdx.x;
  float s = 0.f;
  for (int p = 0; p < SRCLEN; ++p)
    s += enc[((size_t)b * SRCLEN + p) * DD + d];
  ctx[(size_t)b * DD + d] = s * (1.f / (float)SRCLEN);
}

// ---------------- fused logsumexp + top-4 (jax tie-break: low index first) ----
__device__ __forceinline__ void top4_insert(float v, int i, float* tv, int* ti) {
  if (v > tv[3] || (v == tv[3] && i < ti[3])) {
    tv[3] = v; ti[3] = i;
    #pragma unroll
    for (int j = 3; j > 0; --j) {
      if (tv[j] > tv[j-1] || (tv[j] == tv[j-1] && ti[j] < ti[j-1])) {
        float fv = tv[j]; tv[j] = tv[j-1]; tv[j-1] = fv;
        int   fi = ti[j]; ti[j] = ti[j-1]; ti[j-1] = fi;
      } else break;
    }
  }
}

__global__ __launch_bounds__(256) void row_reduce(
    const float* __restrict__ logits, int NV,
    float* __restrict__ topv, int* __restrict__ topi,
    int maskEos, int onlyEos)
{
  const int row = blockIdx.x;
  const int tid = threadIdx.x;
  const float* L = logits + (size_t)row * NV;

  float m = -INFINITY, s = 0.f;
  float tv[4] = {NEGINF, NEGINF, NEGINF, NEGINF};
  int   ti[4] = {0x7fffffff, 0x7fffffff, 0x7fffffff, 0x7fffffff};

  for (int c = tid; c < NV; c += 256) {
    float x = L[c];
    if (x > m) { s = s * __expf(m - x) + 1.f; m = x; }
    else       { s += __expf(x - m); }
    float mv = x;
    if (c == PAD_T)            mv = NEGINF;
    if (maskEos && c == EOS_T) mv = NEGINF;
    if (onlyEos && c != EOS_T) mv = NEGINF;
    top4_insert(mv, c, tv, ti);
  }

  __shared__ float sm[256], ss[256];
  __shared__ float stv[256 * 4];
  __shared__ int   sti[256 * 4];
  sm[tid] = m; ss[tid] = s;
  #pragma unroll
  for (int j = 0; j < 4; ++j) { stv[tid*4+j] = tv[j]; sti[tid*4+j] = ti[j]; }
  __syncthreads();

  for (int off = 128; off > 0; off >>= 1) {
    if (tid < off) {
      float m2 = sm[tid+off], s2 = ss[tid+off];
      if (m2 > sm[tid]) { ss[tid] = ss[tid] * __expf(sm[tid] - m2) + s2; sm[tid] = m2; }
      else              { ss[tid] += s2 * __expf(m2 - sm[tid]); }
    }
    __syncthreads();
  }

  float mtv[4]; int mti[4];
  if (tid < 32) {
    #pragma unroll
    for (int j = 0; j < 4; ++j) { mtv[j] = NEGINF; mti[j] = 0x7fffffff; }
    for (int g = tid * 8; g < tid * 8 + 8; ++g)
      #pragma unroll
      for (int j = 0; j < 4; ++j) top4_insert(stv[g*4+j], sti[g*4+j], mtv, mti);
  }
  __syncthreads();
  if (tid < 32)
    #pragma unroll
    for (int j = 0; j < 4; ++j) { stv[tid*4+j] = mtv[j]; sti[tid*4+j] = mti[j]; }
  __syncthreads();

  if (tid == 0) {
    float ftv[4]; int fti[4];
    #pragma unroll
    for (int j = 0; j < 4; ++j) { ftv[j] = NEGINF; fti[j] = 0x7fffffff; }
    for (int g = 0; g < 32; ++g)
      #pragma unroll
      for (int j = 0; j < 4; ++j) top4_insert(stv[g*4+j], sti[g*4+j], ftv, fti);
    float lsev = logf(ss[0]) + sm[0];
    #pragma unroll
    for (int j = 0; j < 4; ++j) {
      float v = ftv[j];
      topv[(size_t)row * 4 + j] = (v < -1e29f) ? NEGINF : v - lsev;  // lp
      topi[(size_t)row * 4 + j] = fti[j];
    }
  }
}

// ---------------- beam init (after step-0 top-k on 32 rows) ----------------
__global__ __launch_bounds__(64) void beam_init(
    const float* __restrict__ topv, const int* __restrict__ topi,
    int* __restrict__ tok, float* __restrict__ sc, float* __restrict__ fin)
{
  int b = blockIdx.x, tid = threadIdx.x;
  for (int bm = 0; bm < 4; ++bm) {
    int row = b * 4 + bm;
    for (int c = tid; c < TOKW; c += 64) {
      int v = PAD_T;
      if (c == 0) v = EOS_T;
      else if (c == 1) v = topi[b * 4 + bm];
      tok[row * TOKW + c] = v;
    }
    for (int c = tid; c < SCW; c += 64)
      sc[row * SCW + c] = (c == 0) ? topv[b * 4 + bm] : NEGINF;
  }
  if (tid == 0) fin[b] = NEGINF;
}

// ---------------- one beam-search step (per-sentence) ----------------
__global__ __launch_bounds__(64) void beam_step(
    const float* __restrict__ topv, const int* __restrict__ topi,
    const int* __restrict__ tokOld, const float* __restrict__ scOld,
    int* __restrict__ tokNew, float* __restrict__ scNew,
    float* __restrict__ fin, int step)
{
  int b = blockIdx.x, tid = threadIdx.x;
  __shared__ int   sBeamSrc[4], sTok[4];
  __shared__ float sCs[4];

  if (tid == 0) {
    float cs[16], sent[16]; int ct[16];
    float inv = 1.f / (float)(step + 1);
    for (int bm = 0; bm < 4; ++bm) {
      int row = b * 4 + bm;
      float prev = scOld[row * SCW + (step - 1)];
      for (int j = 0; j < 4; ++j) {
        int q = bm * 4 + j;
        cs[q]   = topv[row * 4 + j] + prev;
        ct[q]   = topi[row * 4 + j];
        sent[q] = cs[q] * inv;
      }
    }
    // top-8 of sent (strict > keeps lowest index on ties, matching jax)
    int ord[8]; float sv[8]; bool used[16];
    for (int q = 0; q < 16; ++q) used[q] = false;
    for (int k = 0; k < 8; ++k) {
      int best = 0; float bv = -INFINITY;
      for (int q = 0; q < 16; ++q)
        if (!used[q] && sent[q] > bv) { bv = sent[q]; best = q; }
      used[best] = true; ord[k] = best; sv[k] = bv;
    }
    // finished hypotheses: EOS in first BEAM slots
    float bf = fin[b];
    for (int k = 0; k < 4; ++k)
      if (ct[ord[k]] == EOS_T) bf = fmaxf(bf, sv[k]);
    fin[b] = bf;
    // continue with top-4 non-EOS among the 8
    int sel = 0;
    for (int k = 0; k < 8 && sel < 4; ++k) {
      int q = ord[k];
      if (ct[q] != EOS_T) {
        sBeamSrc[sel] = q >> 2; sTok[sel] = ct[q]; sCs[sel] = cs[q]; ++sel;
      }
    }
    for (; sel < 4; ++sel) { sBeamSrc[sel] = 0; sTok[sel] = PAD_T; sCs[sel] = NEGINF; }
  }
  __syncthreads();

  for (int nb = 0; nb < 4; ++nb) {
    int srcRow = b * 4 + sBeamSrc[nb];
    int dstRow = b * 4 + nb;
    for (int c = tid; c < TOKW; c += 64)
      tokNew[dstRow * TOKW + c] = (c == step + 1) ? sTok[nb] : tokOld[srcRow * TOKW + c];
    for (int c = tid; c < SCW; c += 64)
      scNew[dstRow * SCW + c] = (c == step) ? sCs[nb] : scOld[srcRow * SCW + c];
  }
}

// ---------------- pack outputs ----------------
__global__ __launch_bounds__(256) void final_copy(
    const int* __restrict__ tok, const float* __restrict__ sc,
    const float* __restrict__ fin, float* __restrict__ out)
{
  int i = blockIdx.x * 256 + threadIdx.x;
  const int ntok = NROWS * TOKW;           // 6400
  const int nsc  = NROWS * SCW;            // 6272
  if (i < ntok)                ((int*)out)[i] = tok[i];
  else if (i < ntok + nsc)     out[i] = sc[i - ntok];
  else if (i < ntok + nsc+32)  out[i] = fin[i - ntok - nsc];
}

// ================================================================
extern "C" void kernel_launch(void* const* d_in, const int* in_sizes, int n_in,
                              void* d_out, int out_size, void* d_ws, size_t ws_size,
                              hipStream_t stream) {
  const int*   src   = (const int*)  d_in[0];   // 32*128
  const float* embed = (const float*)d_in[1];   // VOCAB*512
  const float* Wenc  = (const float*)d_in[2];   // 512*512
  const float* Wdec  = (const float*)d_in[3];   // 512*512
  const float* bdec  = (const float*)d_in[4];   // 512

  char* ws = (char*)d_ws;
  size_t off = 0;
  auto carve = [&](size_t bytes) { char* p = ws + off; off += (bytes + 255) & ~size_t(255); return p; };

  __bf16* ebf    = (__bf16*)carve((size_t)VPAD * DD * 2);     // 51.6 MB, zero-padded rows
  __bf16* wenct  = (__bf16*)carve((size_t)DD * DD * 2);
  __bf16* wdect  = (__bf16*)carve((size_t)DD * DD * 2);
  float*  ctx    = (float*) carve((size_t)BSZ * DD * 4);
  __bf16* hbf    = (__bf16*)carve((size_t)NROWS * DD * 2);
  float*  topv   = (float*) carve((size_t)NROWS * 4 * 4);
  int*    topi   = (int*)   carve((size_t)NROWS * 4 * 4);
  float*  fin    = (float*) carve((size_t)BSZ * 4);
  int*    tokA   = (int*)   carve((size_t)NROWS * TOKW * 4);
  int*    tokB   = (int*)   carve((size_t)NROWS * TOKW * 4);
  float*  scA    = (float*) carve((size_t)NROWS * SCW * 4);
  float*  scB    = (float*) carve((size_t)NROWS * SCW * 4);
  float*  logits = (float*) carve((size_t)NROWS * VOCABN * 4); // 25.7 MB
  float*  enc    = logits;  // encoder scratch (8 MB) overlaps logits region

  // ---- prep: bf16 embed (L2-resident), transposed bf16 weights
  {
    int nblk = (int)(((size_t)VPAD * DD + 255) / 256);
    conv_embed<<<nblk, 256, 0, stream>>>(embed, ebf);
    conv_transpose<<<dim3(DD, 2), 256, 0, stream>>>(Wenc, wenct);
    conv_transpose<<<dim3(DD, 2), 256, 0, stream>>>(Wdec, wdect);
  }

  // ---- encoder: enc = tanh(embed[src] @ W_enc), M=4096
  emb_gemm_tanh<<<dim3(DD / 64, 4096 / 16), 128, 0, stream>>>(
      src, 1, 0, 0, ebf, wenct, nullptr, nullptr, 0, enc, nullptr, DD);
  ctx_mean<<<dim3(BSZ, 2), 256, 0, stream>>>(enc, ctx);

  const int ngx = VPAD / 256;   // 197 N-blocks of 256

  // ---- step 0: prev = EOS, M = 32 rows (one per sentence)
  emb_gemm_tanh<<<dim3(DD / 64, 2), 128, 0, stream>>>(
      nullptr, 0, 0, EOS_T, ebf, wdect, bdec, ctx, 0, nullptr, hbf, DD);
  logits_gemm<<<dim3(ngx, 2), 128, 0, stream>>>(hbf, ebf, logits, VOCABN);
  row_reduce<<<BSZ, 256, 0, stream>>>(logits, VOCABN, topv, topi, /*maskEos=*/1, /*onlyEos=*/0);
  beam_init<<<BSZ, 64, 0, stream>>>(topv, topi, tokA, scA, fin);

  // ---- decode loop: steps 1..48 (sequential dependency)
  int* tokCur = tokA; int* tokNxt = tokB;
  float* scCur = scA; float* scNxt = scB;
  for (int step = 1; step <= MAXTGT; ++step) {
    emb_gemm_tanh<<<dim3(DD / 64, NROWS / 16), 128, 0, stream>>>(
        tokCur, TOKW, step, 0, ebf, wdect, bdec, ctx, 2, nullptr, hbf, DD);
    logits_gemm<<<dim3(ngx, NROWS / 16), 128, 0, stream>>>(hbf, ebf, logits, VOCABN);
    row_reduce<<<NROWS, 256, 0, stream>>>(logits, VOCABN, topv, topi, 0,
                                          /*onlyEos=*/(step == MAXTGT) ? 1 : 0);
    beam_step<<<BSZ, 64, 0, stream>>>(topv, topi, tokCur, scCur, tokNxt, scNxt, fin, step);
    int* ti_ = tokCur; tokCur = tokNxt; tokNxt = ti_;
    float* si_ = scCur; scCur = scNxt; scNxt = si_;
  }

  // ---- pack: tokens (int bits), scores, best_fin
  const int ototal = NROWS * TOKW + NROWS * SCW + BSZ;
  final_copy<<<(ototal + 255) / 256, 256, 0, stream>>>(tokCur, scCur, fin, (float*)d_out);
}
